// LePEAttention_30374008718105
// MI455X (gfx1250) — matrix operand order repaired and checked by reading, not compiled, not run on hardware
//
#include <hip/hip_runtime.h>
#include <math.h>

typedef __attribute__((ext_vector_type(16))) _Float16 v16h;
typedef __attribute__((ext_vector_type(8)))  _Float16 v8h;
typedef __attribute__((ext_vector_type(8)))  float    v8f;

#define HEADS_  8
#define HD_     32
#define CDIM    256
#define BATCH   16
#define HIMG    56
#define WIMG    56
#define WSP     7
#define WIN     392      // 56*7 window size
#define NWCOLS  8        // 56/7 windows along W
#define NPIX    3136     // 56*56
#define QPAD    400      // 25*16
#define KPAD    416      // 13*32
#define NQB     25
#define NKB     13

// Load a 16x32 f16 A-operand tile from row-major LDS.
// Layout: lanes 0-15 hold row (lane), K 0-15 contiguous;
//         lanes 16-31 hold row (lane-16), K 16-31 contiguous.
__device__ __forceinline__ v16h load_a16(const _Float16* base, int stride, int lane) {
    int g = lane >> 4;
    int r = lane & 15;
    const _Float16* p = base + r * stride;
    v8h lo = *(const v8h*)(p + g * 16);
    v8h hi = *(const v8h*)(p + g * 16 + 8);
    v16h out;
#pragma unroll
    for (int i = 0; i < 8; ++i) { out[i] = lo[i]; out[8 + i] = hi[i]; }
    return out;
}

__global__ __launch_bounds__(256)
void lepe_attn_kernel(const float* __restrict__ qkv, float* __restrict__ out) {
    const int head = blockIdx.x & 7;
    const int widx = blockIdx.x >> 3;        // 0..127
    const int b    = widx >> 3;
    const int nw   = widx & 7;
    const int tid  = threadIdx.x;
    const int lane = tid & 31;
    const int wave = tid >> 5;
    const int g    = lane >> 4;

    extern __shared__ _Float16 smem[];
    _Float16* sQ  = smem;                 // QPAD x 32 row-major, pre-scaled
    _Float16* sK  = sQ + QPAD * 32;       // KPAD x 32 row-major
    _Float16* sVT = sK + KPAD * 32;       // 32 x KPAD transposed: [d][k]

    const float scale = 0.17677669529663687f;   // 32^-0.5
    const size_t tstride = (size_t)BATCH * NPIX * CDIM;
    const float* qg = qkv + (size_t)b * NPIX * CDIM + (size_t)head * HD_;
    const float* kg = qg + tstride;
    const float* vg = kg + tstride;

    // ---- stage q/k/v for this (window, head) into LDS as f16 ----
    for (int idx = tid; idx < WIN * 8; idx += 256) {
        int row = idx >> 3;
        int seg = idx & 7;
        int h = row / WSP;
        int w = row - h * WSP;
        size_t goff = (size_t)(h * WIMG + nw * WSP + w) * CDIM + seg * 4;
        float4 fq = *(const float4*)(qg + goff);
        float4 fk = *(const float4*)(kg + goff);
        float4 fv = *(const float4*)(vg + goff);
        int d0 = seg * 4;
        _Float16* q4 = sQ + row * 32 + d0;
        q4[0] = (_Float16)(fq.x * scale); q4[1] = (_Float16)(fq.y * scale);
        q4[2] = (_Float16)(fq.z * scale); q4[3] = (_Float16)(fq.w * scale);
        _Float16* k4 = sK + row * 32 + d0;
        k4[0] = (_Float16)fk.x; k4[1] = (_Float16)fk.y;
        k4[2] = (_Float16)fk.z; k4[3] = (_Float16)fk.w;
        sVT[(d0 + 0) * KPAD + row] = (_Float16)fv.x;
        sVT[(d0 + 1) * KPAD + row] = (_Float16)fv.y;
        sVT[(d0 + 2) * KPAD + row] = (_Float16)fv.z;
        sVT[(d0 + 3) * KPAD + row] = (_Float16)fv.w;
    }
    // zero-fill pads
    for (int i = tid; i < (QPAD - WIN) * 32; i += 256) sQ[WIN * 32 + i] = (_Float16)0.f;
    for (int i = tid; i < (KPAD - WIN) * 32; i += 256) sK[WIN * 32 + i] = (_Float16)0.f;
    for (int i = tid; i < 32 * (KPAD - WIN); i += 256) {
        int d = i / (KPAD - WIN);
        int k = i - d * (KPAD - WIN);
        sVT[d * KPAD + WIN + k] = (_Float16)0.f;
    }
    __syncthreads();

    // ---- flash attention: each wave owns q-blocks wave, wave+8, ... ----
    for (int qb = wave; qb < NQB; qb += 8) {
        // B operand Q^T: lane holds column = q-row (lane&15), K(d) chunk g*16..+15 contiguous
        v16h bq = *(const v16h*)(sQ + (qb * 16 + (lane & 15)) * 32 + g * 16);
        v8f o0 = {}; v8f o1 = {};          // O^T accum: d rows 0-15 / 16-31
        float m = -INFINITY, l = 0.f;

        for (int kb = 0; kb < NKB; ++kb) {
            v16h ak0 = load_a16(sK + (kb * 32) * 32, 32, lane);       // k rows +0..15
            v16h ak1 = load_a16(sK + (kb * 32 + 16) * 32, 32, lane);  // k rows +16..31
            v8f zc = {};
            v8f s0 = __builtin_amdgcn_wmma_f32_16x16x32_f16(false, ak0, false, bq, (short)0, zc, false, false);
            v8f s1 = __builtin_amdgcn_wmma_f32_16x16x32_f16(false, ak1, false, bq, (short)0, zc, false, false);

            // C-layout of S^T: element (lane, r) = [k = kb*32 + r + 8g (+16)], q-row = lane&15
            float sv0[8], sv1[8];
            float mloc = -INFINITY;
#pragma unroll
            for (int r = 0; r < 8; ++r) {
                int k0 = kb * 32 + r + 8 * g;
                float a = (k0 < WIN)      ? s0[r] : -INFINITY;
                float c = (k0 + 16 < WIN) ? s1[r] : -INFINITY;
                sv0[r] = a; sv1[r] = c;
                mloc = fmaxf(mloc, fmaxf(a, c));
            }
            mloc = fmaxf(mloc, __shfl_xor(mloc, 16, 32));
            float mnew = fmaxf(m, mloc);
            float corr = __expf(m - mnew);          // 0 on first block (m=-inf)

            float p0[8], p1[8];
            float lloc = 0.f;
#pragma unroll
            for (int r = 0; r < 8; ++r) {
                p0[r] = __expf(sv0[r] - mnew);
                p1[r] = __expf(sv1[r] - mnew);
                lloc += p0[r] + p1[r];
            }
            lloc += __shfl_xor(lloc, 16, 32);
            l = l * corr + lloc;
            m = mnew;
#pragma unroll
            for (int r = 0; r < 8; ++r) { o0[r] *= corr; o1[r] *= corr; }

            // Build B operand P^T: lane column = q-row (lane&15); lanes<16 need k 0..15,
            // lanes>=16 need k 16..31 -> half-wave swap of the two S^T tiles.
            v16h bp;
#pragma unroll
            for (int r = 0; r < 8; ++r) {
                float x0 = __shfl_xor(p0[r], 16, 32);
                float x1 = __shfl_xor(p1[r], 16, 32);
                bp[r]     = (_Float16)((g == 0) ? p0[r] : x1);
                bp[8 + r] = (_Float16)((g == 0) ? x0 : p1[r]);
            }

            v16h av0 = load_a16(sVT + kb * 32, KPAD, lane);            // d rows 0..15
            v16h av1 = load_a16(sVT + 16 * KPAD + kb * 32, KPAD, lane);// d rows 16..31
            o0 = __builtin_amdgcn_wmma_f32_16x16x32_f16(false, av0, false, bp, (short)0, o0, false, false);
            o1 = __builtin_amdgcn_wmma_f32_16x16x32_f16(false, av1, false, bp, (short)0, o1, false, false);
        }

        // O^T C-layout: (lane, r) = q-row lane&15, d = r + 8g (+16); l is per-lane (same q-row)
        int qrow = qb * 16 + (lane & 15);
        if (qrow < WIN) {
            float inv = 1.0f / l;
            int h = qrow / WSP;
            int w = qrow - h * WSP;
            float* op = out + ((size_t)b * NPIX + h * WIMG + nw * WSP + w) * CDIM + head * HD_;
#pragma unroll
            for (int r = 0; r < 8; ++r) {
                int d = r + 8 * g;
                op[d]      = o0[r] * inv;
                op[16 + d] = o1[r] * inv;
            }
        }
    }
}

// LePE: per-window depthwise 3x3 conv over v image (C, 56, 7), zero-padded at
// window edges, added into the attention output.
__global__ __launch_bounds__(256)
void lepe_conv_kernel(const float* __restrict__ qkv, const float* __restrict__ wgt,
                      const float* __restrict__ bias, float* __restrict__ out) {
    size_t idx = (size_t)blockIdx.x * 256 + threadIdx.x;
    int c = (int)(idx & 255);                 // channel fastest -> coalesced
    size_t rest = idx >> 8;
    int p = (int)(rest % WIN);
    int widx = (int)(rest / WIN);
    if (widx >= BATCH * NWCOLS) return;
    int b  = widx >> 3;
    int nw = widx & 7;
    int h = p / WSP;
    int w = p - h * WSP;
    const float* vgb = qkv + 2 * (size_t)BATCH * NPIX * CDIM + (size_t)b * NPIX * CDIM;
    float sum = bias[c];
#pragma unroll
    for (int dy = -1; dy <= 1; ++dy) {
        int hh = h + dy;
        if (hh < 0 || hh >= HIMG) continue;
#pragma unroll
        for (int dx = -1; dx <= 1; ++dx) {
            int ww = w + dx;
            if (ww < 0 || ww >= WSP) continue;   // window-local zero padding
            int n = hh * WIMG + nw * WSP + ww;
            sum += vgb[(size_t)n * CDIM + c] * wgt[c * 9 + (dy + 1) * 3 + (dx + 1)];
        }
    }
    int n0 = h * WIMG + nw * WSP + w;
    out[((size_t)b * NPIX + n0) * CDIM + c] += sum;
}

extern "C" void kernel_launch(void* const* d_in, const int* in_sizes, int n_in,
                              void* d_out, int out_size, void* d_ws, size_t ws_size,
                              hipStream_t stream) {
    const float* qkv = (const float*)d_in[0];
    const float* lw  = (const float*)d_in[1];
    const float* lb  = (const float*)d_in[2];
    float* out = (float*)d_out;

    size_t shmem = (size_t)(QPAD * 32 + KPAD * 32 + 32 * KPAD) * 2; // f16 bytes = 78848
    lepe_attn_kernel<<<dim3(BATCH * NWCOLS * HEADS_), dim3(256), shmem, stream>>>(qkv, out);

    int total = BATCH * NWCOLS * WIN * CDIM;   // 12,845,056
    lepe_conv_kernel<<<dim3(total / 256), dim3(256), 0, stream>>>(qkv, lw, lb, out);
}
